// FeedForward_12421045420273
// MI455X (gfx1250) — compile-verified, matrix operand order checked by reading, not run on hardware
//
#include <hip/hip_runtime.h>
#include <hip/hip_bf16.h>

// ---------------------------------------------------------------------------
// SpikingJelly-style FeedForward block for MI455X (gfx1250, wave32, WMMA).
//   s1 = LIF(x); h = s1 @ W1^T; h = BN1(h); s2 = LIF(h); y = s2 @ W2^T;
//   out = BN2(y) + x
// Shapes: T=4, B=32, C=512, Ch=2048, H=W=14 (HW=196).
//
// Roofline: 105 GFLOP of GEMM vs ~154 MB of unavoidable HBM traffic (6.6 us
// at 23.3 TB/s); all intermediates (~133 MB fp16) are L2-resident (192 MB), so
// the matrix pipe is the bottleneck -> f16 WMMA (spikes are exactly {0,1}).
//
// Key optimization this round: batch the T=4 timesteps in registers inside
// GEMM1 (their GEMMs are independent; only the LIF2 recursion in the epilogue
// couples them), and 64-row M-blocking in GEMM2. Both reach 16 WMMAs per
// 16 global_load_b128 (1.0 load/WMMA vs 2.5 before), and weights are loaded
// once per k-step instead of once per timestep.
// ---------------------------------------------------------------------------

typedef __attribute__((ext_vector_type(16))) _Float16 v16h;
typedef __attribute__((ext_vector_type(8)))  _Float16 v8h;
typedef __attribute__((ext_vector_type(8)))  float    v8f;

#define TT 4
#define BB 32
#define CC 512
#define CH 2048
#define HWX 196
#define NP (BB * HWX)   // 6272 rows per timestep (b*196+hw)
#define NTOT (TT * NP)  // 25088 total GEMM rows

// ---------------------------------------------------------------------------
// Fragment loader: row-major [rows, ld] fp16 matrix -> 16x32 (or 32x16) WMMA
// fragment per CDNA5 ISA layout: lane L holds row (L&15); K-halves at
// k0+(L>>4)*8 .. +8 and k0+16+(L>>4)*8 .. +8  (two 16B loads).
// ---------------------------------------------------------------------------
__device__ __forceinline__ v16h load_frag(const _Float16* __restrict__ p,
                                          int row, int k0, int ld, int lane) {
  const _Float16* q = p + (size_t)(row + (lane & 15)) * ld + (k0 + ((lane >> 4) << 3));
  v8h lo = *(const v8h*)q;
  v8h hi = *(const v8h*)(q + 16);
  v16h r;
#pragma unroll
  for (int i = 0; i < 8; ++i) { r[i] = lo[i]; r[8 + i] = hi[i]; }
  return r;
}

// ---------------------------------------------------------------------------
// fp32 -> fp16 weight conversion
// ---------------------------------------------------------------------------
__global__ void cvt_f32_to_f16_kernel(const float* __restrict__ src,
                                      _Float16* __restrict__ dst, int n) {
  int i = blockIdx.x * blockDim.x + threadIdx.x;
  if (i < n) dst[i] = (_Float16)src[i];
}

// ---------------------------------------------------------------------------
// Fold BN: scale = gamma * rsqrt(var+eps); bias = beta - mean*scale
// ---------------------------------------------------------------------------
__global__ void bnprep_kernel(const float* __restrict__ g, const float* __restrict__ be,
                              const float* __restrict__ m, const float* __restrict__ va,
                              float* __restrict__ sc, float* __restrict__ bi, int n) {
  int i = blockIdx.x * blockDim.x + threadIdx.x;
  if (i < n) {
    float inv = g[i] * rsqrtf(va[i] + 1e-5f);
    sc[i] = inv;
    bi[i] = be[i] - m[i] * inv;
  }
}

// ---------------------------------------------------------------------------
// LIF over input + layout transpose.
// in : fp32 [T,B,C,HW]  ->  s1: fp16 [t*NP + b*196+hw, c]  (GEMM row-major)
// One thread per (n'=b*196+hw, c); t-scan in registers. Stores coalesced in c.
// ---------------------------------------------------------------------------
__global__ void lif1_transpose_kernel(const float* __restrict__ x,
                                      _Float16* __restrict__ s1) {
  int i = blockIdx.x * blockDim.x + threadIdx.x;  // i = n'*512 + c
  if (i >= NP * CC) return;
  int c  = i & (CC - 1);
  int np = i >> 9;
  int b  = np / HWX;
  int hw = np - b * HWX;
  const size_t inbase = ((size_t)b * CC + c) * HWX + hw;
  float v = 0.0f;
#pragma unroll
  for (int t = 0; t < TT; ++t) {
    float xin = x[inbase + (size_t)t * BB * CC * HWX];
    v = 0.5f * (v + xin);                  // charge: v += (x - v)/tau, tau=2
    float s = (v >= 1.0f) ? 1.0f : 0.0f;   // fire (VTH = 1)
    v = v * (1.0f - s);                    // hard reset (detached)
    s1[(size_t)(t * NP + np) * CC + c] = (_Float16)s;
  }
}

// ---------------------------------------------------------------------------
// GEMM1 (s1 @ W1^T) + BN1 + LIF2, fused, T-batched.
// Each wave owns a 16x64 tile of the (NP x Ch) space for ALL 4 timesteps at
// once: 16 accumulator tiles (t x j). Per k-step: 4 A frags + 4 B frags ->
// 16 WMMAs (weights read once, not once per t). The LIF2 recursion runs
// sequentially over t in the epilogue, with membrane state in the same
// D-fragment registers (lane = N%16 + 16*(M>=8), VGPR r = M%8).
// grid = (NP/16, Ch/512), block = 256 (8 waves x 64 cols).
// ---------------------------------------------------------------------------
__global__ void __launch_bounds__(256, 1)
gemm1_bn_lif2_kernel(const _Float16* __restrict__ s1,
                     const _Float16* __restrict__ w1,   // [Ch, C] row-major fp16
                     const float* __restrict__ sc, const float* __restrict__ bi,
                     _Float16* __restrict__ s2) {       // [NTOT, Ch] fp16
  const int lane = threadIdx.x & 31;
  const int wave = threadIdx.x >> 5;
  const int m0 = blockIdx.x * 16;
  const int n0 = blockIdx.y * 512 + wave * 64;
  const int mlane = m0 + (lane >> 4) * 8;

  v8f vzero = {};
  v8f acc[TT][4];
#pragma unroll
  for (int t = 0; t < TT; ++t)
#pragma unroll
    for (int j = 0; j < 4; ++j) acc[t][j] = vzero;

  for (int k0 = 0; k0 < CC; k0 += 32) {
    v16h a[TT];
#pragma unroll
    for (int t = 0; t < TT; ++t)
      a[t] = load_frag(s1, t * NP + m0, k0, CC, lane);
#pragma unroll
    for (int j = 0; j < 4; ++j) {
      v16h bf = load_frag(w1, n0 + j * 16, k0, CC, lane);
#pragma unroll
      for (int t = 0; t < TT; ++t)
        acc[t][j] = __builtin_amdgcn_wmma_f32_16x16x32_f16(
            false, a[t], false, bf, (short)0, acc[t][j], false, false);
    }
  }

  // epilogue: per column tile, scan t sequentially: BN1 -> LIF2 -> spike store
  v8f vst[4];
#pragma unroll
  for (int j = 0; j < 4; ++j) vst[j] = vzero;

#pragma unroll
  for (int t = 0; t < TT; ++t) {
#pragma unroll
    for (int j = 0; j < 4; ++j) {
      int c = n0 + j * 16 + (lane & 15);
      float s_ = sc[c], b_ = bi[c];
#pragma unroll
      for (int r = 0; r < 8; ++r) {
        float h = acc[t][j][r] * s_ + b_;
        float v = vst[j][r];
        v = 0.5f * (v + h);
        float sp = (v >= 1.0f) ? 1.0f : 0.0f;
        vst[j][r] = v * (1.0f - sp);
        s2[(size_t)(t * NP + mlane + r) * CH + c] = (_Float16)sp;
      }
    }
  }
}

// ---------------------------------------------------------------------------
// GEMM2 (s2 @ W2^T) + BN2 + residual add, fp32 out in original [T,B,C,HW].
// 64x64 tile per wave (4 row-groups x 4 col tiles = 16 accumulators);
// per k-step: 4 A frags + 4 B frags -> 16 WMMAs.
// grid = (NTOT/64), block = 256 (8 waves x 64 cols cover all 512 cols).
// ---------------------------------------------------------------------------
__global__ void __launch_bounds__(256, 1)
gemm2_bn_res_kernel(const _Float16* __restrict__ s2,
                    const _Float16* __restrict__ w2,    // [C, Ch] row-major fp16
                    const float* __restrict__ sc, const float* __restrict__ bi,
                    const float* __restrict__ xin,      // fp32 [T,B,C,HW]
                    float* __restrict__ out) {          // fp32 [T,B,C,HW]
  const int lane = threadIdx.x & 31;
  const int wave = threadIdx.x >> 5;
  const int m0 = blockIdx.x * 64;
  const int n0 = wave * 64;

  v8f vzero = {};
  v8f acc[4][4];
#pragma unroll
  for (int g = 0; g < 4; ++g)
#pragma unroll
    for (int j = 0; j < 4; ++j) acc[g][j] = vzero;

  for (int k0 = 0; k0 < CH; k0 += 32) {
    v16h a[4];
#pragma unroll
    for (int g = 0; g < 4; ++g)
      a[g] = load_frag(s2, m0 + g * 16, k0, CH, lane);
#pragma unroll
    for (int j = 0; j < 4; ++j) {
      v16h bf = load_frag(w2, n0 + j * 16, k0, CH, lane);
#pragma unroll
      for (int g = 0; g < 4; ++g)
        acc[g][j] = __builtin_amdgcn_wmma_f32_16x16x32_f16(
            false, a[g], false, bf, (short)0, acc[g][j], false, false);
    }
  }

#pragma unroll
  for (int g = 0; g < 4; ++g) {
    int mlane = m0 + g * 16 + (lane >> 4) * 8;
#pragma unroll
    for (int j = 0; j < 4; ++j) {
      int c = n0 + j * 16 + (lane & 15);
      float s_ = sc[c], b_ = bi[c];
#pragma unroll
      for (int r = 0; r < 8; ++r) {
        int n = mlane + r;              // global GEMM row = t*NP + b*196 + hw
        int t  = n / NP;
        int rr = n - t * NP;
        int b  = rr / HWX;
        int hw = rr - b * HWX;
        size_t adr = (((size_t)t * BB + b) * CC + c) * HWX + hw;
        out[adr] = acc[g][j][r] * s_ + b_ + xin[adr];
      }
    }
  }
}

// ---------------------------------------------------------------------------
// Host-side launcher
// ---------------------------------------------------------------------------
extern "C" void kernel_launch(void* const* d_in, const int* in_sizes, int n_in,
                              void* d_out, int out_size, void* d_ws, size_t ws_size,
                              hipStream_t stream) {
  (void)in_sizes; (void)n_in; (void)out_size; (void)ws_size;

  const float* x      = (const float*)d_in[0];
  const float* W1     = (const float*)d_in[1];
  const float* gamma1 = (const float*)d_in[2];
  const float* beta1  = (const float*)d_in[3];
  const float* mean1  = (const float*)d_in[4];
  const float* var1   = (const float*)d_in[5];
  const float* W2     = (const float*)d_in[6];
  const float* gamma2 = (const float*)d_in[7];
  const float* beta2  = (const float*)d_in[8];
  const float* mean2  = (const float*)d_in[9];
  const float* var2   = (const float*)d_in[10];
  float* out = (float*)d_out;

  // Workspace carve-up (256B aligned): ~132.7 MB total, L2-resident.
  char* ws = (char*)d_ws;
  size_t off = 0;
  _Float16* s1  = (_Float16*)(ws + off); off += (size_t)NTOT * CC * 2;   // 25.7 MB
  _Float16* w1h = (_Float16*)(ws + off); off += (size_t)CH * CC * 2;     //  2.1 MB
  _Float16* w2h = (_Float16*)(ws + off); off += (size_t)CC * CH * 2;     //  2.1 MB
  _Float16* s2  = (_Float16*)(ws + off); off += (size_t)NTOT * CH * 2;   // 102.8 MB
  float* bn1s = (float*)(ws + off); off += CH * 4;
  float* bn1b = (float*)(ws + off); off += CH * 4;
  float* bn2s = (float*)(ws + off); off += CC * 4;
  float* bn2b = (float*)(ws + off); off += CC * 4;

  // 1) weights fp32 -> fp16 (layout [O,K] already B-fragment-friendly)
  {
    int n = CH * CC;
    cvt_f32_to_f16_kernel<<<(n + 255) / 256, 256, 0, stream>>>(W1, w1h, n);
    cvt_f32_to_f16_kernel<<<(n + 255) / 256, 256, 0, stream>>>(W2, w2h, n);
  }

  // 2) fold BN constants
  bnprep_kernel<<<(CH + 255) / 256, 256, 0, stream>>>(gamma1, beta1, mean1, var1,
                                                      bn1s, bn1b, CH);
  bnprep_kernel<<<(CC + 255) / 256, 256, 0, stream>>>(gamma2, beta2, mean2, var2,
                                                      bn2s, bn2b, CC);

  // 3) LIF over input + transpose to GEMM layout (spikes as fp16)
  {
    int n = NP * CC;
    lif1_transpose_kernel<<<(n + 255) / 256, 256, 0, stream>>>(x, s1);
  }

  // 4) GEMM1 + BN1 + LIF2 fused, T-batched (weights read once per k-step)
  {
    dim3 grid(NP / 16, CH / 512);   // (392, 4)
    gemm1_bn_lif2_kernel<<<grid, 256, 0, stream>>>(s1, w1h, bn1s, bn1b, s2);
  }

  // 5) GEMM2 + BN2 + residual, 64x64 tiles per wave
  {
    dim3 grid(NTOT / 64, 1);        // (392, 1)
    gemm2_bn_res_kernel<<<grid, 256, 0, stream>>>(s2, w2h, bn2s, bn2b, x, out);
  }
}
